// populationEncodedBinaryClassifier_40690520162410
// MI455X (gfx1250) — compile-verified, matrix operand order checked by reading, not run on hardware
//
#include <hip/hip_runtime.h>

// ---------------------------------------------------------------------------
// LIF spiking layer, MI455X (gfx1250), wave32 + WMMA bf16 + LDS-staged A.
//   T=100, B=512, I=2048, O=256
//   Weff[o,i] = W[o,i]^2 + nW[o,i]   (precomputed, split into bf16 hi+lo,
//                                     packed in WMMA B-fragment lane order)
//   per t:  cur = x_t @ Weff^T  via 3-product bf16 WMMA (fp32-class accuracy)
//           mem += cur; spk = mem > 1; mem *= (1-spk)
//   out = concat(spk_rec[T,B,O], mem_rec[T,B,O])   (float32)
//
// Main kernel geometry:
//   workgroup = 128 threads = 4 waves; all waves share the same 16 batch rows
//   wave w owns N-tiles {ntile0, ntile0+1} (16 rows x 32 cols of output)
//   grid = (B/16, 2) = (32, 2); N-half per blockIdx.y
//   K staged in LDS chunks of 128 (4 fragments of 32); wave w converts
//   fragment w (f32 -> bf16 hi/lo, WMMA A-fragment lane order) cooperatively.
// Workspace: 2 MB packed weights.
// ---------------------------------------------------------------------------

typedef __attribute__((ext_vector_type(16))) __bf16 v16bf;
typedef __attribute__((ext_vector_type(8)))  float  v8f;

#define T_STEPS 100
#define BATCH   512
#define N_IN    2048
#define N_OUT   256
#define K_ITERS (N_IN / 32)          // 64 WMMA k-steps of 32
#define N_TILES (N_OUT / 16)         // 16 column tiles of 16
#define KC_FRAGS 4                   // 32-k fragments per LDS chunk (1/wave)
#define KCHUNKS (K_ITERS / KC_FRAGS) // 16 chunks of K=128 per timestep
#define FRAG_HALVES 512              // 32 lanes * 16 halves per fragment
#define PACK_ELEMS ((size_t)N_TILES * K_ITERS * FRAG_HALVES) // 524288 = 1MB

__device__ __forceinline__ unsigned short bf16_rne(float f) {
    unsigned u = __float_as_uint(f);
    unsigned r = u + 0x7FFFu + ((u >> 16) & 1u);   // round-to-nearest-even
    return (unsigned short)(r >> 16);
}
__device__ __forceinline__ float bf16_up(unsigned short h) {
    return __uint_as_float(((unsigned)h) << 16);
}

// ---------------------------------------------------------------------------
// Kernel 1 (one-time): Weff = W*W + nW, bf16 hi/lo, packed per-lane in the
// V_WMMA_F32_16X16X32_BF16 B-fragment order:
//   lane L: column n = ntile*16 + (L&15); kgrp = (L<16) ? 0 : 8
//   halves h=0..7 -> K = kgrp + h ; h=8..15 -> K = kgrp + 16 + (h-8)
// ---------------------------------------------------------------------------
__global__ void lif_prep_weights(const float* __restrict__ W,
                                 const float* __restrict__ nW,
                                 unsigned short* __restrict__ whi,
                                 unsigned short* __restrict__ wlo) {
    int gid  = blockIdx.x * blockDim.x + threadIdx.x;
    int lane = gid & 31;
    int blk  = gid >> 5;                 // ntile*K_ITERS + kit
    if (blk >= N_TILES * K_ITERS) return;
    int kit   = blk % K_ITERS;
    int ntile = blk / K_ITERS;
    int n     = ntile * 16 + (lane & 15);
    int kgrp  = (lane & 16) ? 8 : 0;
    size_t base = (size_t)blk * FRAG_HALVES + (size_t)lane * 16;
    #pragma unroll
    for (int h = 0; h < 16; ++h) {
        int klocal = (h < 8) ? (kgrp + h) : (kgrp + 16 + (h - 8));
        size_t idx = (size_t)n * N_IN + (size_t)kit * 32 + klocal;
        float w    = W[idx];
        float nw   = nW[idx];
        float weff = w * w + nw;
        unsigned short hi = bf16_rne(weff);
        whi[base + h] = hi;
        wlo[base + h] = bf16_rne(weff - bf16_up(hi));
    }
}

// ---------------------------------------------------------------------------
// Kernel 2: T-persistent WMMA LIF scan with LDS-staged A fragments.
// ---------------------------------------------------------------------------
__global__ void __launch_bounds__(128)
lif_wmma_scan(const float* __restrict__ x,
              const unsigned short* __restrict__ whi,
              const unsigned short* __restrict__ wlo,
              float* __restrict__ out) {
    __shared__ __align__(16) __bf16 Ah[KC_FRAGS * FRAG_HALVES];  // 4 KB
    __shared__ __align__(16) __bf16 Al[KC_FRAGS * FRAG_HALVES];  // 4 KB

    const int lane   = threadIdx.x & 31;
    const int wave   = threadIdx.x >> 5;                 // 0..3
    const int m0     = blockIdx.x * 16;                  // batch-row group
    const int ntile0 = blockIdx.y * 8 + wave * 2;        // first of 2 N-tiles
    const int crow   = m0 + (lane & 15);                 // conversion src row
    const int g0     = (lane & 16) ? 8 : 0;              // A-frag k swizzle
    const int mb     = (lane & 16) ? 8 : 0;              // C-layout row base
    const int abase  = wave * FRAG_HALVES + lane * 16;   // LDS write slot
    const size_t TBO = (size_t)T_STEPS * BATCH * N_OUT;

    v8f mem[2] = {};

    for (int t = 0; t < T_STEPS; ++t) {
        // this thread converts fragment `wave` of each chunk from row `crow`
        const float* xrow = x + ((size_t)t * BATCH + crow) * N_IN + wave * 32 + g0;
        v8f acc[2] = {};

        for (int c = 0; c < KCHUNKS; ++c) {
            // ---- cooperative convert: 16 f32 -> bf16 hi/lo into LDS --------
            const float* xa = xrow + c * (KC_FRAGS * 32);
            float4 a0 = *(const float4*)(xa + 0);
            float4 a1 = *(const float4*)(xa + 4);
            float4 a2 = *(const float4*)(xa + 16);
            float4 a3 = *(const float4*)(xa + 20);
            if (c + 1 < KCHUNKS) __builtin_prefetch(xa + KC_FRAGS * 32, 0, 0);

            float av[16] = { a0.x, a0.y, a0.z, a0.w,  a1.x, a1.y, a1.z, a1.w,
                             a2.x, a2.y, a2.z, a2.w,  a3.x, a3.y, a3.z, a3.w };
            #pragma unroll
            for (int h = 0; h < 16; ++h) {
                __bf16 hb = (__bf16)av[h];               // native RNE cvt
                Ah[abase + h] = hb;
                Al[abase + h] = (__bf16)(av[h] - (float)hb);
            }
            __syncthreads();

            // ---- compute: 4 fragments x 2 N-tiles x 3-product WMMA ---------
            #pragma unroll
            for (int f = 0; f < KC_FRAGS; ++f) {
                union { v16bf v; float4 q[2]; } XH, XL;
                const float4* ph = (const float4*)&Ah[f * FRAG_HALVES + lane * 16];
                XH.q[0] = ph[0]; XH.q[1] = ph[1];
                const float4* pl = (const float4*)&Al[f * FRAG_HALVES + lane * 16];
                XL.q[0] = pl[0]; XL.q[1] = pl[1];

                const int kit = c * KC_FRAGS + f;
                #pragma unroll
                for (int ti = 0; ti < 2; ++ti) {
                    size_t boff = ((size_t)(ntile0 + ti) * K_ITERS + kit) * FRAG_HALVES
                                + (size_t)lane * 16;
                    union { v16bf v; float4 q[2]; } BH, BL;
                    const float4* bh = (const float4*)(whi + boff);
                    BH.q[0] = bh[0]; BH.q[1] = bh[1];
                    const float4* bl = (const float4*)(wlo + boff);
                    BL.q[0] = bl[0]; BL.q[1] = bl[1];

                    acc[ti] = __builtin_amdgcn_wmma_f32_16x16x32_bf16(
                        false, XH.v, false, BH.v, (short)0, acc[ti], false, false);
                    acc[ti] = __builtin_amdgcn_wmma_f32_16x16x32_bf16(
                        false, XL.v, false, BH.v, (short)0, acc[ti], false, false);
                    acc[ti] = __builtin_amdgcn_wmma_f32_16x16x32_bf16(
                        false, XH.v, false, BL.v, (short)0, acc[ti], false, false);
                }
            }
            __syncthreads();
        }

        // ---- LIF update + record (C layout: VGPR v -> row m0+mb+v) --------
        #pragma unroll
        for (int ti = 0; ti < 2; ++ti) {
            const int coln = (ntile0 + ti) * 16 + (lane & 15);
            #pragma unroll
            for (int v = 0; v < 8; ++v) {
                float m = mem[ti][v] + acc[ti][v];
                float spk = (m > 1.0f) ? 1.0f : 0.0f;
                m = (spk != 0.0f) ? 0.0f : m;             // reset-to-zero
                mem[ti][v] = m;
                size_t o = ((size_t)t * BATCH + (m0 + mb + v)) * N_OUT + coln;
                out[o]       = spk;                       // spk_rec
                out[TBO + o] = m;                         // mem_rec (post-reset)
            }
        }
    }
}

extern "C" void kernel_launch(void* const* d_in, const int* in_sizes, int n_in,
                              void* d_out, int out_size, void* d_ws, size_t ws_size,
                              hipStream_t stream) {
    const float* x  = (const float*)d_in[0];   // [100, 512, 2048]
    const float* W  = (const float*)d_in[1];   // [256, 2048]
    const float* nW = (const float*)d_in[2];   // [256, 2048]
    float* out = (float*)d_out;                // [2, 100, 512, 256] flat

    unsigned short* whi = (unsigned short*)d_ws;           // 1 MB
    unsigned short* wlo = whi + PACK_ELEMS;                // +1 MB

    // Kernel 1: pack effective weights (16*64 fragments * 32 lanes)
    int prep_threads = N_TILES * K_ITERS * 32;
    lif_prep_weights<<<(prep_threads + 255) / 256, 256, 0, stream>>>(W, nW, whi, wlo);

    // Kernel 2: grid = 32 batch-row groups x 2 N-halves, 128 threads (4 waves)
    dim3 grid(BATCH / 16, 2);
    lif_wmma_scan<<<grid, 128, 0, stream>>>(x, whi, wlo, out);
}
